// HPFA_70471823393385
// MI455X (gfx1250) — compile-verified, matrix operand order checked by reading, not run on hardware
//
#include <hip/hip_runtime.h>
#include <hip/hip_bf16.h>
#include <math.h>

typedef _Float16 half_t;
typedef __attribute__((ext_vector_type(16))) _Float16 v16h;
typedef __attribute__((ext_vector_type(8)))  _Float16 v8h;
typedef __attribute__((ext_vector_type(8)))  float    v8f;

union FragH { v16h v; v8h g[2]; _Float16 h[16]; };
union AccU  { v8f v; float f[8]; };

__device__ __forceinline__ v8h zero8() {
    v8h z; for (int i = 0; i < 8; i++) z[i] = (_Float16)0; return z;
}

__device__ __forceinline__ v8f wmma16(v16h a, v16h b, v8f c) {
    // D = A(16x32 f16) x B(32x16 f16) + C(16x16 f32)
    return __builtin_amdgcn_wmma_f32_16x16x32_f16(false, a, false, b, (short)0, c, false, false);
}

// Load a 16x32 A-fragment (or B-fragment from B^T) from a row-major [rows x K] array.
// Per CDNA5 f16 A layout: lane<16 holds K 0..7,16..23; lane>=16 holds K 8..15,24..31.
// kvalid (multiple of 8) zero-pads the contraction dimension.
__device__ __forceinline__ v16h load_frag(const half_t* base, int ld, int kvalid) {
    int lane = threadIdx.x & 31;
    const half_t* p = base + (size_t)(lane & 15) * ld;
    int kb = (lane & 16) ? 8 : 0;
    FragH u;
    u.g[0] = (kb < kvalid)      ? *(const v8h*)(p + kb)      : zero8();
    u.g[1] = (kb + 16 < kvalid) ? *(const v8h*)(p + kb + 16) : zero8();
    return u.v;
}

// Same but only rows 0..7 are valid (used for the 8-row tok_w2 A operand).
__device__ __forceinline__ v16h load_frag_rows8(const half_t* base, int ld) {
    int lane = threadIdx.x & 31;
    int row = lane & 15;
    int kb = (lane & 16) ? 8 : 0;
    FragH u;
    if (row < 8) {
        const half_t* p = base + (size_t)row * ld;
        u.g[0] = *(const v8h*)(p + kb);
        u.g[1] = *(const v8h*)(p + kb + 16);
    } else { u.g[0] = zero8(); u.g[1] = zero8(); }
    return u.v;
}

__device__ __forceinline__ float gelu_f(float v) {
    return 0.5f * v * (1.0f + erff(v * 0.70710678118654752f));
}

// ---------------------------------------------------------------- converters
__global__ void k_cvt(const float* __restrict__ s, half_t* __restrict__ d, int n) {
    int i = blockIdx.x * blockDim.x + threadIdx.x;
    if (i < n) d[i] = (half_t)s[i];
}

// -------------------------------------------------- A1: means -> Y (8192,136) f16
__global__ void k_means(const float* __restrict__ x, half_t* __restrict__ Yh) {
    int wave = (blockIdx.x * blockDim.x + threadIdx.x) >> 5;
    int lane = threadIdx.x & 31;
    const float* xb = x + (size_t)wave * 1024;
    float cs0 = 0, cs1 = 0, cs2 = 0, cs3 = 0;
    float rowp[8];
    for (int r = 0; r < 8; r++) {
        float4 v = *(const float4*)(xb + r * 128 + lane * 4);
        rowp[r] = v.x + v.y + v.z + v.w;
        cs0 += v.x; cs1 += v.y; cs2 += v.z; cs3 += v.w;
    }
    for (int r = 0; r < 8; r++) {
        float s = rowp[r];
        for (int o = 16; o >= 1; o >>= 1) s += __shfl_xor(s, o);
        rowp[r] = s;
    }
    half_t* yb = Yh + (size_t)wave * 136;
    if (lane == 0)
        for (int r = 0; r < 8; r++) yb[r] = (half_t)(rowp[r] * (1.0f / 128.0f));
    typedef __attribute__((ext_vector_type(4))) _Float16 v4h;
    v4h cw;
    cw[0] = (half_t)(cs0 * 0.125f); cw[1] = (half_t)(cs1 * 0.125f);
    cw[2] = (half_t)(cs2 * 0.125f); cw[3] = (half_t)(cs3 * 0.125f);
    *(v4h*)(yb + 8 + lane * 4) = cw;
}

// --------------------- A2: Y @ sun_w^T, BN, hard-swish, sigmoid -> s (8192,128) f32
__global__ __launch_bounds__(256) void k_sun(
        const half_t* __restrict__ Yh, const half_t* __restrict__ Wh,
        const float* __restrict__ cb, const float* __restrict__ bg,
        const float* __restrict__ bb, const float* __restrict__ bm,
        const float* __restrict__ bv, float* __restrict__ s_out) {
    int wave = (blockIdx.x * blockDim.x + threadIdx.x) >> 5;
    int lane = threadIdx.x & 31;
    int l15 = lane & 15, hi = (lane >> 4) & 1;
    int m0 = wave * 16;
    v16h afr[5];
    for (int kc = 0; kc < 5; kc++) {
        int k0 = kc * 32, kv = (k0 + 32 <= 136) ? 32 : (136 - k0);
        afr[kc] = load_frag(Yh + (size_t)m0 * 136 + k0, 136, kv);
    }
    for (int t = 0; t < 8; t++) {
        AccU acc; for (int j = 0; j < 8; j++) acc.f[j] = 0.f;
        for (int kc = 0; kc < 5; kc++) {
            int k0 = kc * 32, kv = (k0 + 32 <= 136) ? 32 : (136 - k0);
            v16h bfr = load_frag(Wh + (size_t)(16 * t) * 136 + k0, 136, kv);
            acc.v = wmma16(afr[kc], bfr, acc.v);
        }
        int n = 16 * t + l15;
        float sc = bg[n] * rsqrtf(bv[n] + 1e-5f);
        float off = bb[n] - bm[n] * sc;
        float cbn = cb[n];
        for (int j = 0; j < 8; j++) {
            float z = (acc.f[j] + cbn) * sc + off;
            float hs = z * fminf(fmaxf(z + 3.f, 0.f), 6.f) * (1.f / 6.f);
            s_out[(size_t)(m0 + j + 8 * hi) * 128 + n] = 1.f / (1.f + __expf(-hs));
        }
    }
}

// ------------------------------------- A3: fused MLP-mixer, 2 batches per wave
__global__ __launch_bounds__(256) void k_mixer(
        const float* __restrict__ x, const float* __restrict__ s,
        const half_t* __restrict__ w1t, const float* __restrict__ b1t,
        const half_t* __restrict__ w2t, const float* __restrict__ b2t,
        const half_t* __restrict__ w1c, const float* __restrict__ b1c,
        const half_t* __restrict__ w2c, const float* __restrict__ b2c,
        float* __restrict__ hout) {
    __shared__ half_t lds[8 * 2560];           // per wave: 2048 Hrm + 512 Grm
    int wid = threadIdx.x >> 5;
    int lane = threadIdx.x & 31;
    int l15 = lane & 15, hi = (lane >> 4) & 1;
    half_t* Hrm = lds + wid * 2560;
    half_t* Grm = Hrm + 2048;
    size_t pair = (size_t)blockIdx.x * 8 + wid;
    int myb = (int)(pair * 2) + hi;            // lane-high half holds batch1

    // ---- init: h = x * a_w[c] * a_h[r], kept f32 in D-fragment layout ----
    const float* sb = s + (size_t)myb * 128;
    float4 a0 = *(const float4*)sb, a1 = *(const float4*)(sb + 4);
    float ahv[8] = {a0.x, a0.y, a0.z, a0.w, a1.x, a1.y, a1.z, a1.w};
    float Hf[8][8];
    for (int t = 0; t < 8; t++) {
        int c = 16 * t + l15;
        float aw = sb[c];
        const float* xb = x + (size_t)myb * 1024 + c;
        for (int j = 0; j < 8; j++) Hf[t][j] = xb[j * 128] * aw * ahv[j];
    }
    float4 r0 = *(const float4*)b2t, r1 = *(const float4*)(b2t + 4);
    float b2v[8] = {r0.x, r0.y, r0.z, r0.w, r1.x, r1.y, r1.z, r1.w};

    for (int iter = 0; iter < 2; iter++) {
        // ================= token mixing: h += (W2 gelu(W1 h + b1) + b2) =====
        for (int bb = 0; bb < 2; bb++) {
            v16h Bt[8];                        // B = H (K=8 rows x N=128), zero-padded K
            for (int t = 0; t < 8; t++) {
                FragH u;
                for (int e = 0; e < 8; e++) {
                    float hv = __shfl(Hf[t][e], l15 + 16 * bb);
                    u.h[e]     = (lane < 16) ? (half_t)hv : (half_t)0;
                    u.h[e + 8] = (half_t)0;
                }
                Bt[t] = u.v;
            }
            AccU macc[8];
            for (int t = 0; t < 8; t++) for (int j = 0; j < 8; j++) macc[t].f[j] = 0.f;
            for (int fc = 0; fc < 16; fc++) {
                int f0 = fc * 32;
                if (fc + 1 < 16) {             // pull next weight chunk toward the WGP
                    __builtin_prefetch(w1t + (size_t)(f0 + 32) * 8 + l15 * 16, 0, 3);
                    __builtin_prefetch(w2t + (size_t)l15 * 512 + f0 + 32, 0, 3);
                }
                v16h A0 = load_frag(w1t + (size_t)f0 * 8, 8, 8);
                v16h A1 = load_frag(w1t + (size_t)(f0 + 16) * 8, 8, 8);
                v16h A2 = load_frag_rows8(w2t + f0, 512);
                const float* bp = b1t + f0 + 8 * hi;
                float4 q0 = *(const float4*)bp,        q1 = *(const float4*)(bp + 4);
                float4 q2 = *(const float4*)(bp + 16), q3 = *(const float4*)(bp + 20);
                float b1a[8] = {q0.x, q0.y, q0.z, q0.w, q1.x, q1.y, q1.z, q1.w};
                float b1b[8] = {q2.x, q2.y, q2.z, q2.w, q3.x, q3.y, q3.z, q3.w};
                for (int t = 0; t < 8; t++) {
                    AccU d0, d1;
                    for (int j = 0; j < 8; j++) { d0.f[j] = 0.f; d1.f[j] = 0.f; }
                    d0.v = wmma16(A0, Bt[t], d0.v);
                    d1.v = wmma16(A1, Bt[t], d1.v);
                    // D-pair is exactly the next WMMA's B k-pattern: pack in-register
                    FragH gb;
                    for (int e = 0; e < 8; e++) {
                        gb.h[e]     = (half_t)gelu_f(d0.f[e] + b1a[e]);
                        gb.h[e + 8] = (half_t)gelu_f(d1.f[e] + b1b[e]);
                    }
                    macc[t].v = wmma16(A2, gb.v, macc[t].v);
                }
            }
            for (int t = 0; t < 8; t++)
                for (int j = 0; j < 8; j++) {
                    float v = __shfl(macc[t].f[j], l15);   // broadcast valid low half
                    if (hi == bb) Hf[t][j] += v + b2v[j];
                }
        }
        // stage h (f16, row-major 16x128) for channel A-fragments (wave-private LDS)
        for (int t = 0; t < 8; t++)
            for (int j = 0; j < 8; j++)
                Hrm[(j + 8 * hi) * 128 + 16 * t + l15] = (half_t)Hf[t][j];

        // ====== channel mixing: h += gelu(h W1^T + b1) W2^T + b2  (streamed K) ======
        v16h Ach[4];
        for (int ks = 0; ks < 4; ks++) Ach[ks] = load_frag(Hrm + 32 * ks, 128, 32);
        AccU cacc[8];
        for (int t = 0; t < 8; t++) for (int j = 0; j < 8; j++) cacc[t].f[j] = Hf[t][j];
        for (int nc = 0; nc < 16; nc++) {
            int n0 = nc * 32;
            if (nc + 1 < 16) {
                __builtin_prefetch(w1c + (size_t)(n0 + 32 + l15) * 128, 0, 3);
                __builtin_prefetch(w2c + (size_t)l15 * 512 + n0 + 32, 0, 3);
            }
            for (int tt = 0; tt < 2; tt++) {
                AccU d; for (int j = 0; j < 8; j++) d.f[j] = 0.f;
                for (int ks = 0; ks < 4; ks++) {
                    v16h bfr = load_frag(w1c + (size_t)(n0 + 16 * tt) * 128 + 32 * ks, 128, 32);
                    d.v = wmma16(Ach[ks], bfr, d.v);
                }
                float b1v = b1c[n0 + 16 * tt + l15];
                for (int j = 0; j < 8; j++)
                    Grm[(j + 8 * hi) * 32 + 16 * tt + l15] = (half_t)gelu_f(d.f[j] + b1v);
            }
            v16h A2c = load_frag(Grm, 32, 32);       // same-wave LDS: in-order, no barrier
            for (int t = 0; t < 8; t++) {
                v16h bfr = load_frag(w2c + (size_t)(16 * t) * 512 + n0, 512, 32);
                cacc[t].v = wmma16(A2c, bfr, cacc[t].v);
            }
        }
        for (int t = 0; t < 8; t++) {
            float bv = b2c[16 * t + l15];
            for (int j = 0; j < 8; j++) Hf[t][j] = cacc[t].f[j] + bv;
        }
    }
    for (int t = 0; t < 8; t++) {
        int c = 16 * t + l15;
        float* ob = hout + (size_t)myb * 1024 + c;
        for (int j = 0; j < 8; j++) ob[j * 128] = Hf[t][j];
    }
}

// ------------------------------------------------ B1: desc reduction (64,128)
__global__ void k_reduce(const float* __restrict__ m0, const float* __restrict__ m1,
                         const float* __restrict__ m2, float* __restrict__ desc) {
    int chunk = blockIdx.x;      // 0..13 block-rows across modules (8+4+2)
    int b = blockIdx.y;
    const float* src;
    if (chunk < 8)       src = m0 + ((size_t)b * 8 + chunk)       * 128 * 512;
    else if (chunk < 12) src = m1 + ((size_t)b * 4 + (chunk - 8))  * 128 * 512;
    else                 src = m2 + ((size_t)b * 2 + (chunk - 12)) * 128 * 512;
    int wid = threadIdx.x >> 5, lane = threadIdx.x & 31;
    for (int c = wid; c < 128; c += 8) {
        const float* row = src + (size_t)c * 512;
        float sum = 0.f;
        for (int q = 0; q < 4; q++) {
            float4 v = *(const float4*)(row + 4 * (lane + 32 * q));
            sum += v.x + v.y + v.z + v.w;
        }
        for (int o = 16; o >= 1; o >>= 1) sum += __shfl_xor(sum, o);
        if (lane == 0) atomicAdd(desc + b * 128 + c, sum);
    }
}

// --------------------------- B2: g = relu(bn(desc@jw^T)); att = 0.2+0.8*softmax
__global__ void k_att(const float* __restrict__ desc, const float* __restrict__ jw,
                      const float* __restrict__ jb, const float* __restrict__ jg,
                      const float* __restrict__ jbb, const float* __restrict__ jm,
                      const float* __restrict__ jv, const float* __restrict__ hw,
                      const float* __restrict__ hb, float* __restrict__ att) {
    int b = blockIdx.x;
    __shared__ float g[32];
    __shared__ float dsc[128];
    int t = threadIdx.x;
    dsc[t] = desc[b * 128 + t] * (1.f / 512.f);
    __syncthreads();
    if (t < 32) {
        float a = jb[t];
        for (int d = 0; d < 128; d++) a += dsc[d] * jw[t * 128 + d];
        float sc = jg[t] * rsqrtf(jv[t] + 1e-5f);
        g[t] = fmaxf((a - jm[t]) * sc + jbb[t], 0.f);
    }
    __syncthreads();
    float scv[14];
    float mx = -1e30f;
    for (int k = 0; k < 14; k++) {
        float a = hb[k * 128 + t];
        const float* w = hw + ((size_t)k * 128 + t) * 32;
        for (int d = 0; d < 32; d++) a += g[d] * w[d];
        scv[k] = a; mx = fmaxf(mx, a);
    }
    float se = 0.f;
    for (int k = 0; k < 14; k++) { scv[k] = __expf(scv[k] - mx); se += scv[k]; }
    float inv = 0.8f / se;
    for (int k = 0; k < 14; k++)
        att[((size_t)b * 14 + k) * 128 + t] = 0.2f + scv[k] * inv;
}

// -------------------- B3: out_i = m_i * att (division-free, streaming float4)
__global__ __launch_bounds__(256) void k_scale(
        const float4* __restrict__ m0, const float4* __restrict__ m1,
        const float4* __restrict__ m2, const float* __restrict__ att,
        float4* __restrict__ o0, float4* __restrict__ o1,
        float4* __restrict__ o2) {
    int chunk = blockIdx.x;      // 0..13 block-rows across modules (8+4+2)
    int b = blockIdx.y;
    const float4* src; float4* dst;
    if (chunk < 8) {
        size_t off = ((size_t)b * 8 + chunk) * 16384;        // 128 rows * 128 f4
        src = m0 + off; dst = o0 + off;
    } else if (chunk < 12) {
        size_t off = ((size_t)b * 4 + (chunk - 8)) * 16384;
        src = m1 + off; dst = o1 + off;
    } else {
        size_t off = ((size_t)b * 2 + (chunk - 12)) * 16384;
        src = m2 + off; dst = o2 + off;
    }
    __shared__ float sa[128];
    if (threadIdx.x < 128)
        sa[threadIdx.x] = att[((size_t)b * 14 + chunk) * 128 + threadIdx.x];
    __syncthreads();
    for (int idx = threadIdx.x; idx < 16384; idx += 256) {
        float a = sa[idx >> 7];
        float4 v = src[idx];
        v.x *= a; v.y *= a; v.z *= a; v.w *= a;
        dst[idx] = v;
    }
}

extern "C" void kernel_launch(void* const* d_in, const int* in_sizes, int n_in,
                              void* d_out, int out_size, void* d_ws, size_t ws_size,
                              hipStream_t stream) {
    (void)in_sizes; (void)n_in; (void)out_size; (void)ws_size;
    const float* x    = (const float*)d_in[0];
    const float* m0   = (const float*)d_in[1];
    const float* m1   = (const float*)d_in[2];
    const float* m2   = (const float*)d_in[3];
    const float* sunw = (const float*)d_in[4];
    const float* sunb = (const float*)d_in[5];
    const float* sbg  = (const float*)d_in[6];
    const float* sbb  = (const float*)d_in[7];
    const float* sbm  = (const float*)d_in[8];
    const float* sbv  = (const float*)d_in[9];
    const float* tw1  = (const float*)d_in[10];
    const float* tb1  = (const float*)d_in[11];
    const float* tw2  = (const float*)d_in[12];
    const float* tb2  = (const float*)d_in[13];
    const float* cw1  = (const float*)d_in[14];
    const float* cb1  = (const float*)d_in[15];
    const float* cw2  = (const float*)d_in[16];
    const float* cb2  = (const float*)d_in[17];
    const float* jw   = (const float*)d_in[18];
    const float* jb   = (const float*)d_in[19];
    const float* jg   = (const float*)d_in[20];
    const float* jbb  = (const float*)d_in[21];
    const float* jm   = (const float*)d_in[22];
    const float* jv   = (const float*)d_in[23];
    const float* hw   = (const float*)d_in[24];
    const float* hb   = (const float*)d_in[25];

    char* w = (char*)d_ws;
    auto take = [&](size_t bytes) -> char* {
        char* p = w; w += (bytes + 255) & ~(size_t)255; return p;
    };
    half_t* sunh = (half_t*)take((size_t)136 * 136 * 2);
    half_t* tw1h = (half_t*)take((size_t)4096 * 2);
    half_t* tw2h = (half_t*)take((size_t)4096 * 2);
    half_t* cw1h = (half_t*)take((size_t)65536 * 2);
    half_t* cw2h = (half_t*)take((size_t)65536 * 2);
    half_t* Yh   = (half_t*)take((size_t)8192 * 136 * 2);
    float*  sws  = (float*)take((size_t)8192 * 128 * 4);
    float*  desc = (float*)take((size_t)64 * 128 * 4);
    float*  att  = (float*)take((size_t)64 * 14 * 128 * 4);

    float* hout = (float*)d_out;
    float* o0 = hout + (size_t)8192 * 8 * 128;
    float* o1 = o0 + (size_t)64 * 1024 * 512;
    float* o2 = o1 + (size_t)64 * 512 * 512;

    // weight conversion to f16
    k_cvt<<<(18496 + 255) / 256, 256, 0, stream>>>(sunw, sunh, 18496);
    k_cvt<<<16, 256, 0, stream>>>(tw1, tw1h, 4096);
    k_cvt<<<16, 256, 0, stream>>>(tw2, tw2h, 4096);
    k_cvt<<<256, 256, 0, stream>>>(cw1, cw1h, 65536);
    k_cvt<<<256, 256, 0, stream>>>(cw2, cw2h, 65536);
    // Part A
    k_means<<<1024, 256, 0, stream>>>(x, Yh);
    k_sun<<<64, 256, 0, stream>>>(Yh, sunh, sunb, sbg, sbb, sbm, sbv, sws);
    k_mixer<<<512, 256, 0, stream>>>(x, sws, tw1h, tb1, tw2h, tb2,
                                     cw1h, cb1, cw2h, cb2, hout);
    // Part B
    hipMemsetAsync(desc, 0, (size_t)64 * 128 * 4, stream);
    k_reduce<<<dim3(14, 64), 256, 0, stream>>>(m0, m1, m2, desc);
    k_att<<<64, 128, 0, stream>>>(desc, jw, jb, jg, jbb, jm, jv, hw, hb, att);
    k_scale<<<dim3(14, 64), 256, 0, stream>>>((const float4*)m0, (const float4*)m1,
                                              (const float4*)m2, att,
                                              (float4*)o0, (float4*)o1, (float4*)o2);
}
